// Decode_One_Level_89850715832348
// MI455X (gfx1250) — matrix operand order repaired and checked by reading, not compiled, not run on hardware
//
#include <hip/hip_runtime.h>

#define Bn 4
#define Cc 96
#define Hh 128
#define Ww 256
#define HW (Hh*Ww)
#define FCH 433      // feat channels (output)
#define MCH 448      // bf16 mirror channels (padded to /32)

typedef __attribute__((ext_vector_type(8)))  unsigned short v8u;
typedef __attribute__((ext_vector_type(16))) unsigned short v16u;
typedef __attribute__((ext_vector_type(16))) __bf16         v16bf;
typedef __attribute__((ext_vector_type(8)))  float          v8f;

union BC16 { v16u u; v16bf b; };

__device__ __forceinline__ unsigned short f2bf(float f) {
  unsigned u = __float_as_uint(f);
  u += 0x7FFFu + ((u >> 16) & 1u);          // round-to-nearest-even
  return (unsigned short)(u >> 16);
}

// ---------------------------------------------------------------------------
// Correlation (81 channels) + leaky ReLU -> feat[352..432] fp32 + bf16 mirror
// grid (H, 81, B), block = W (256)
// ---------------------------------------------------------------------------
__global__ __launch_bounds__(256) void corr_kernel(
    const float* __restrict__ a, const float* __restrict__ bsec,
    float* __restrict__ feat, unsigned short* __restrict__ mir)
{
  const int x = threadIdx.x;
  const int y = blockIdx.x;
  const int d = blockIdx.y;
  const int b = blockIdx.z;
  const int dy = d / 9 - 4, dx = d % 9 - 4;
  const int yy = y + dy, xx = x + dx;
  float acc = 0.f;
  if ((unsigned)yy < (unsigned)Hh && (unsigned)xx < (unsigned)Ww) {
    const float* ap = a    + (size_t)b*Cc*HW + y*Ww + x;
    const float* bp = bsec + (size_t)b*Cc*HW + yy*Ww + xx;
    #pragma unroll 8
    for (int c = 0; c < Cc; ++c) acc = fmaf(ap[c*HW], bp[c*HW], acc);
  }
  acc = (acc >= 0.f) ? acc : 0.1f*acc;
  feat[((size_t)(b*FCH + 352 + d))*HW + y*Ww + x] = acc;
  mir[((size_t)((b*Hh + y)*Ww + x))*MCH + 352 + d] = f2bf(acc);
}

// ---------------------------------------------------------------------------
// Pack conv weights fp32 [Cout][Cin][3][3] into per-lane B-fragment order:
//   Wp[((tap*nChunks+chunk)*nTiles+tile)*512 + lane*16 + j]
//   value = bf16( W[co=tile*16+lane%16][ci=chunk*32 + ((j&7) + 8*(lane>=16) + 16*(j>=8))][tap] )
// zero-padded for co>=Cout or ci>=Cin.
// ---------------------------------------------------------------------------
__global__ void pack_weights(const float* __restrict__ W, unsigned short* __restrict__ Wp,
                             int Cout, int Cin, int nChunks, int nTiles)
{
  const int idx = blockIdx.x * 256 + threadIdx.x;
  const int total = 9 * nChunks * nTiles * 512;
  if (idx >= total) return;
  const int j    = idx & 15;
  const int lane = (idx >> 4) & 31;
  int rest = idx >> 9;
  const int t     = rest % nTiles; rest /= nTiles;
  const int chunk = rest % nChunks;
  const int tap   = rest / nChunks;
  const int hi = lane >> 4, ln = lane & 15;
  const int K  = (j & 7) + 8*hi + 16*(j >> 3); // B operand: K = 8*hi+j (j<8), 16+8*hi+(j-8) (j>=8)
  const int co = t*16 + ln;
  const int ci = chunk*32 + K;
  float v = (co < Cout && ci < Cin) ? W[((size_t)co*Cin + ci)*9 + tap] : 0.f;
  Wp[idx] = f2bf(v);
}

// ---------------------------------------------------------------------------
// Implicit-GEMM 3x3 conv via v_wmma_f32_16x16x32_bf16.
//   M = 16 pixels (A from LDS, documented 16-bit A layout),
//   N = Cout (B = pre-packed weights, contiguous per lane),
//   K = 9 taps * 32-channel chunks.
// One wave owns 16 px * NT*16 cout; 8 waves/WG share the staged input tile.
// MODE 0: lrelu + write feat slice + bf16 mirror. MODE 1: fused flow attention.
// grid (W/128, H, B), block 256.
// ---------------------------------------------------------------------------
template<int NT, int MODE>
__global__ __launch_bounds__(256) void conv_kernel(
    unsigned short* __restrict__ mir,
    const unsigned short* __restrict__ Wp,
    const float* __restrict__ bias,
    float* __restrict__ feat,
    float* __restrict__ flow,
    int inOff, int nChunks, int outOff, int Cout)
{
  __shared__ __align__(16) unsigned short sA[3*136*32];   // [row][x][ci] bf16
  const int tid  = threadIdx.x;
  const int lane = tid & 31, w = tid >> 5;
  const int hi = lane >> 4, ln = lane & 15;
  const int x0 = blockIdx.x * 128;
  const int y  = blockIdx.y;
  const int b  = blockIdx.z;

  v8f acc[NT];
  #pragma unroll
  for (int t = 0; t < NT; ++t) {
    const int co = t*16 + ln;
    const float bv = (co < Cout) ? bias[co] : 0.f;
    #pragma unroll
    for (int i = 0; i < 8; ++i) acc[t][i] = bv;
  }

  const v16u* Wp16 = (const v16u*)Wp;

  for (int ch = 0; ch < nChunks; ++ch) {
    __syncthreads();
    const int cbase = inOff + ch*32;
    // stage 3 rows x 136 px x 32 ci (64B per pixel, contiguous in mirror)
    for (int it = tid; it < 3*136; it += 256) {
      const int row = it / 136, xi = it - row*136;
      const int yy = y + row - 1;
      const int xx = x0 - 4 + xi;
      uint4* dst = (uint4*)(sA + (row*136 + xi)*32);
      if ((unsigned)yy < (unsigned)Hh && (unsigned)xx < (unsigned)Ww) {
        const uint4* src = (const uint4*)(mir + ((size_t)((b*Hh + yy)*Ww + xx))*MCH + cbase);
        dst[0] = src[0]; dst[1] = src[1]; dst[2] = src[2]; dst[3] = src[3];
      } else {
        uint4 z = make_uint4(0u,0u,0u,0u);
        dst[0] = z; dst[1] = z; dst[2] = z; dst[3] = z;
      }
    }
    __syncthreads();

    #pragma unroll
    for (int tap = 0; tap < 9; ++tap) {
      const int kh = tap / 3, kw = tap % 3;
      // A fragment: lane row M = ln -> pixel x0 + w*16 + ln (+kw-1 shift, +4 halo)
      const unsigned short* ap = sA + (kh*136 + (w*16 + ln + kw + 3))*32;
      // 16-bit A layout: lanes<16: K = 0..7,16..23 ; lanes>=16: K = 8..15,24..31
      v8u a0 = *(const v8u*)(ap + 8*hi);
      v8u a1 = *(const v8u*)(ap + 16 + 8*hi);
      BC16 A;
      A.u = __builtin_shufflevector(a0, a1, 0,1,2,3,4,5,6,7,8,9,10,11,12,13,14,15);
      const int wbase = ((tap*nChunks + ch)*NT)*32 + lane;
      #pragma unroll
      for (int t = 0; t < NT; ++t) {
        BC16 Bm; Bm.u = Wp16[wbase + t*32];
        acc[t] = __builtin_amdgcn_wmma_f32_16x16x32_bf16(
            false, A.b, false, Bm.b, (short)0, acc[t], false, false);
      }
    }
  }

  if (MODE == 0) {
    #pragma unroll
    for (int t = 0; t < NT; ++t) {
      const int col = t*16 + ln;
      if (col < Cout) {
        const int co = outOff + col;
        #pragma unroll
        for (int r = 0; r < 8; ++r) {           // C layout: M = r + 8*hi, N = ln
          const int x = x0 + w*16 + r + 8*hi;
          float v = acc[t][r];
          v = (v >= 0.f) ? v : 0.1f*v;
          feat[((size_t)(b*FCH + co))*HW + y*Ww + x] = v;
          mir[((size_t)((b*Hh + y)*Ww + x))*MCH + co] = f2bf(v);
        }
      }
    }
  } else {
    // fused flow attention: mean over 81 channels of sim * {vx,vy}
    #pragma unroll
    for (int r = 0; r < 8; ++r) {
      float sx = 0.f, sy = 0.f;
      #pragma unroll
      for (int t = 0; t < NT; ++t) {
        const int c = t*16 + ln;
        if (c < 81) {
          const float v = acc[t][r];
          sx += v * (float)(c % 9 - 4);
          sy += v * (float)(c / 9 - 4);
        }
      }
      #pragma unroll
      for (int m = 1; m < 16; m <<= 1) {        // butterfly within half-wave
        sx += __shfl_xor(sx, m, 32);
        sy += __shfl_xor(sy, m, 32);
      }
      if (ln == 0) {
        const int x = x0 + w*16 + r + 8*hi;
        flow[((size_t)(b*2 + 0))*HW + y*Ww + x] = sx * (1.f/81.f);
        flow[((size_t)(b*2 + 1))*HW + y*Ww + x] = sy * (1.f/81.f);
      }
    }
  }
}

// ---------------------------------------------------------------------------
extern "C" void kernel_launch(void* const* d_in, const int* in_sizes, int n_in,
                              void* d_out, int out_size, void* d_ws, size_t ws_size,
                              hipStream_t stream)
{
  (void)in_sizes; (void)n_in; (void)out_size;
  const float* tenFirst  = (const float*)d_in[0];
  const float* tenSecond = (const float*)d_in[1];
  const float* Wc[6] = {(const float*)d_in[2],(const float*)d_in[4],(const float*)d_in[6],
                        (const float*)d_in[8],(const float*)d_in[10],(const float*)d_in[12]};
  const float* bc[6] = {(const float*)d_in[3],(const float*)d_in[5],(const float*)d_in[7],
                        (const float*)d_in[9],(const float*)d_in[11],(const float*)d_in[13]};

  float* flow = (float*)d_out;
  float* feat = flow + (size_t)Bn*2*HW;

  static const int cfgCin[6]    = { 81,177,273,337,401,433};
  static const int cfgCout[6]   = { 96, 96, 64, 64, 32, 81};
  static const int cfgChunks[6] = {  3,  6,  9, 11, 13, 14};  // ceil(CinPad/32)
  static const int cfgInOff[6]  = {352,256,160, 96, 32,  0};
  static const int cfgOutOff[6] = {256,160, 96, 32,  0,  0};
  static const int cfgTiles[6]  = {  6,  6,  4,  4,  2,  6};

  const size_t mirHalves = (size_t)Bn*HW*MCH;
  unsigned short* mir = (unsigned short*)d_ws;
  unsigned short* wp[6];
  unsigned short* p = mir + mirHalves;
  size_t need = mirHalves * 2;
  for (int i = 0; i < 6; ++i) {
    wp[i] = p;
    const size_t n = (size_t)9 * cfgChunks[i] * cfgTiles[i] * 512;
    p += n; need += n * 2;
  }
  if (ws_size < need) return;

  // zero mirror (also clears pad channels 433..447) — capturable async memset
  hipMemsetAsync(d_ws, 0, mirHalves * 2, stream);

  for (int i = 0; i < 6; ++i) {
    const int total = 9 * cfgChunks[i] * cfgTiles[i] * 512;
    pack_weights<<<(total + 255)/256, 256, 0, stream>>>(
        Wc[i], wp[i], cfgCout[i], cfgCin[i], cfgChunks[i], cfgTiles[i]);
  }

  corr_kernel<<<dim3(Hh, 81, Bn), Ww, 0, stream>>>(tenFirst, tenSecond, feat, mir);

  const dim3 cg(Ww/128, Hh, Bn);
  conv_kernel<6,0><<<cg,256,0,stream>>>(mir, wp[0], bc[0], feat, flow, cfgInOff[0], cfgChunks[0], cfgOutOff[0], cfgCout[0]);
  conv_kernel<6,0><<<cg,256,0,stream>>>(mir, wp[1], bc[1], feat, flow, cfgInOff[1], cfgChunks[1], cfgOutOff[1], cfgCout[1]);
  conv_kernel<4,0><<<cg,256,0,stream>>>(mir, wp[2], bc[2], feat, flow, cfgInOff[2], cfgChunks[2], cfgOutOff[2], cfgCout[2]);
  conv_kernel<4,0><<<cg,256,0,stream>>>(mir, wp[3], bc[3], feat, flow, cfgInOff[3], cfgChunks[3], cfgOutOff[3], cfgCout[3]);
  conv_kernel<2,0><<<cg,256,0,stream>>>(mir, wp[4], bc[4], feat, flow, cfgInOff[4], cfgChunks[4], cfgOutOff[4], cfgCout[4]);
  conv_kernel<6,1><<<cg,256,0,stream>>>(mir, wp[5], bc[5], feat, flow, cfgInOff[5], cfgChunks[5], cfgOutOff[5], cfgCout[5]);
}
